// Attention_30992484008598
// MI455X (gfx1250) — compile-verified, matrix operand order checked by reading, not run on hardware
//
#include <hip/hip_runtime.h>
#include <hip/hip_bf16.h>
#include <math.h>

// ---------------------------------------------------------------------------
// MHA forward for MI455X (gfx1250, wave32, WMMA).
// f16 data / f32 accumulate via v_wmma_f32_16x16x32_f16; flash attention so
// the S x S score matrix never exists in memory. ~71 GFLOP vs ~50 MB HBM ->
// compute bound; everything rides the matrix pipe.
//
// Round-3: softmax row reductions switched from ds_bpermute shuffles to pure
// VALU DPP16 butterflies (quad_perm xor1/xor2, row_half_mirror, row_mirror)
// so they co-execute with WMMA instead of serializing on s_wait_dscnt; exp
// moved to base-2 (exp2f) with the 1/sqrt(d)*log2(e) constant pre-folded.
// ---------------------------------------------------------------------------

typedef _Float16 f16;
typedef __attribute__((ext_vector_type(16))) _Float16 v16h;
typedef __attribute__((ext_vector_type(8)))  _Float16 v8h;
typedef __attribute__((ext_vector_type(8)))  float    v8f;

#define S_SEQ   4096
#define DMODEL  768
#define NHEADS  12
#define DHEAD   64

// ---------------------------------------------------------------------------
// Fragment helpers for V_WMMA_F32_16X16X32_F16 (ISA 7.12.2 layouts).
// 16-bit A (16x32 MxK): lane l holds row m=l&15; half=l>>4 selects K-bands:
//   elements 0..7  <- K = half*8 + 0..7
//   elements 8..15 <- K = 16 + half*8 + 0..7
// B (32x16 KxN) uses the same striping with N in place of M, so loading a
// row-major N x K buffer (Bt) with this loader yields the B fragment.
// ---------------------------------------------------------------------------
__device__ __forceinline__ v16h load_frag_rowmajor(const f16* __restrict__ p,
                                                   int ld, int lane) {
  const int m    = lane & 15;
  const int half = lane >> 4;
  const f16* r0  = p + (size_t)m * ld + half * 8;
  v8h lo = *(const v8h*)(r0);        // K = half*8 .. half*8+7   (b128 load)
  v8h hi = *(const v8h*)(r0 + 16);   // K = 16+half*8 .. +7      (b128 load)
  v16h out;
#pragma unroll
  for (int i = 0; i < 8; ++i) { out[i] = lo[i]; out[i + 8] = hi[i]; }
  return out;
}

__device__ __forceinline__ v8f wmma_f16(v16h a, v16h b, v8f c) {
  return __builtin_amdgcn_wmma_f32_16x16x32_f16(false, a, false, b,
                                                (short)0, c, false, false);
}

// ---------------------------------------------------------------------------
// DPP16 16-lane row reductions (pure VALU, no LDS round-trip).
// quad_perm(1,0,3,2)=0x0B1, quad_perm(2,3,0,1)=0x04E,
// row_half_mirror=0x141, row_mirror=0x140. Rows of 16 lanes stay independent,
// matching the C-fragment's half-wave row groups.
// ---------------------------------------------------------------------------
template <int CTRL>
__device__ __forceinline__ float dpp_movf(float v) {
  return __builtin_bit_cast(
      float, __builtin_amdgcn_update_dpp(0, __builtin_bit_cast(int, v), CTRL,
                                         0xf, 0xf, true));
}
__device__ __forceinline__ float row16_max(float v) {
  v = fmaxf(v, dpp_movf<0x0B1>(v));
  v = fmaxf(v, dpp_movf<0x04E>(v));
  v = fmaxf(v, dpp_movf<0x141>(v));
  v = fmaxf(v, dpp_movf<0x140>(v));
  return v;
}
__device__ __forceinline__ float row16_sum(float v) {
  v += dpp_movf<0x0B1>(v);
  v += dpp_movf<0x04E>(v);
  v += dpp_movf<0x141>(v);
  v += dpp_movf<0x140>(v);
  return v;
}

// ---------------------------------------------------------------------------
// Cast / transpose-cast preprocessing
// ---------------------------------------------------------------------------
__global__ void cast_f32_to_f16(const float* __restrict__ in,
                                f16* __restrict__ out, int n) {
  int i = blockIdx.x * blockDim.x + threadIdx.x;
  if (i < n) out[i] = (f16)in[i];
}

// W [R][C] row-major f32 -> Wt [C][R] f16 (LDS-tiled transpose, 32x32).
__global__ void transpose_cast_f32_to_f16(const float* __restrict__ W,
                                          f16* __restrict__ Wt, int R, int C) {
  __shared__ float tile[32][33];
  const int c0 = blockIdx.x * 32;
  const int r0 = blockIdx.y * 32;
  const int tx = threadIdx.x;   // 0..31
  const int ty = threadIdx.y;   // 0..7
#pragma unroll
  for (int i = 0; i < 32; i += 8)
    tile[ty + i][tx] = W[(size_t)(r0 + ty + i) * C + (c0 + tx)];
  __syncthreads();
#pragma unroll
  for (int i = 0; i < 32; i += 8)
    Wt[(size_t)(c0 + ty + i) * R + (r0 + tx)] = (f16)tile[tx][ty + i];
}

// ---------------------------------------------------------------------------
// WMMA GEMM: C[M,N] = A[M,K] * Bt[N,K]^T + bias
//   MODE 0: f16 row-major out
//   MODE 1: f16 head-major out      [h][m][64]   (h=col/64, d=col%64)
//   MODE 2: f32 row-major out       (final projection into d_out)
//   MODE 3: f16 head-major TRANSPOSED out [h][64][S]  (for V)
// One 16x16 tile per wave, 8 waves per block -> 16x128 block tile.
// ---------------------------------------------------------------------------
template <int MODE>
__global__ __launch_bounds__(256) void
gemm_wmma(const f16* __restrict__ A, const f16* __restrict__ Bt,
          const float* __restrict__ bias, void* __restrict__ Cout,
          int M, int N, int K) {
  const int lane = threadIdx.x & 31;
  const int wave = threadIdx.x >> 5;
  const int n0 = (blockIdx.x * 8 + wave) * 16;
  const int m0 = blockIdx.y * 16;
  if (n0 >= N) return;

  const f16* a = A + (size_t)m0 * K;
  const f16* b = Bt + (size_t)n0 * K;

  v8f acc = {};
#pragma unroll 2
  for (int k = 0; k < K; k += 32) {
    v16h af = load_frag_rowmajor(a + k, K, lane);
    v16h bf = load_frag_rowmajor(b + k, K, lane);
    acc = wmma_f16(af, bf, acc);
  }

  // C/D layout: lane -> col n = lane&15; element r -> row (lane>>4)*8 + r
  const int n  = lane & 15;
  const int mo = (lane >> 4) * 8;
  const float bval = bias ? bias[n0 + n] : 0.0f;
#pragma unroll
  for (int r = 0; r < 8; ++r) {
    const float v    = acc[r] + bval;
    const size_t row = (size_t)(m0 + mo + r);
    const int    col = n0 + n;
    if constexpr (MODE == 0) {
      ((f16*)Cout)[row * N + col] = (f16)v;
    } else if constexpr (MODE == 1) {
      const int h = col >> 6, d = col & 63;
      ((f16*)Cout)[(((size_t)h * M + row) << 6) + d] = (f16)v;
    } else if constexpr (MODE == 3) {
      // transposed per head: Vt[h][d][s]
      ((f16*)Cout)[(size_t)col * M + row] = (f16)v;   // col = h*64+d
    } else {
      ((float*)Cout)[row * N + col] = v;
    }
  }
}

// ---------------------------------------------------------------------------
// Flash attention: one wave per (head, 16-row Q tile). Streams K/V in 32-row
// chunks; online softmax in registers (DPP16 row reductions, base-2 exp);
// P bounced through LDS to convert the C-fragment layout into an A-fragment
// for the P*V WMMAs. V is pre-transposed ([h][64][S]) so its B fragments are
// plain contiguous b128 loads.
// ---------------------------------------------------------------------------
__global__ __launch_bounds__(32, 1) void
flash_attn_wmma(const f16* __restrict__ Q, const f16* __restrict__ Kh,
                const f16* __restrict__ VhT, f16* __restrict__ Y, int S) {
  __shared__ __align__(16) f16 pbuf[16 * 32];

  const int lane = threadIdx.x & 31;
  const int h    = blockIdx.y;
  const int m0   = blockIdx.x * 16;

  const f16* qh = Q   + ((size_t)h * S + m0) * DHEAD;
  const f16* kt = Kh  + (size_t)h * S * DHEAD;      // running K-tile ptr
  const f16* vt = VhT + (size_t)h * DHEAD * S;      // running Vt-tile ptr

  // Q tile (16x64) as two A fragments (K=0..31, K=32..63), resident all loop.
  const v16h q0 = load_frag_rowmajor(qh,      DHEAD, lane);
  const v16h q1 = load_frag_rowmajor(qh + 32, DHEAD, lane);

  v8f o0 = {}, o1 = {}, o2 = {}, o3 = {};
  float mrow[8], lrow[8];
#pragma unroll
  for (int r = 0; r < 8; ++r) { mrow[r] = -INFINITY; lrow[r] = 0.0f; }

  // base-2 softmax: logits scaled by 1/sqrt(64) * log2(e)
  const float c2 = 0.125f * 1.44269504088896340736f;
  const int n  = lane & 15;
  const int mo = (lane >> 4) * 8;

  for (int j = 0; j < S; j += 32, kt += 32 * DHEAD, vt += 32) {
    // ---- scores: Q (16x64) * K_tile^T -> two 16x16 tiles (4 WMMAs) ----
    v8f s0 = {}, s1 = {};
    {
      v16h b00 = load_frag_rowmajor(kt, DHEAD, lane);
      v16h b01 = load_frag_rowmajor(kt + 32, DHEAD, lane);
      v16h b10 = load_frag_rowmajor(kt + 16 * DHEAD, DHEAD, lane);
      v16h b11 = load_frag_rowmajor(kt + 16 * DHEAD + 32, DHEAD, lane);
      s0 = wmma_f16(q0, b00, s0);
      s0 = wmma_f16(q1, b01, s0);
      s1 = wmma_f16(q0, b10, s1);
      s1 = wmma_f16(q1, b11, s1);
    }

    // ---- online softmax (base-2) over the 32 new columns ----
    float p0[8], p1[8], alpha[8];
#pragma unroll
    for (int r = 0; r < 8; ++r) {
      const float a0 = s0[r] * c2;
      const float a1 = s1[r] * c2;
      const float mx = row16_max(fmaxf(a0, a1));
      const float mn = fmaxf(mrow[r], mx);
      alpha[r] = exp2f(mrow[r] - mn);
      mrow[r]  = mn;
      p0[r] = exp2f(a0 - mn);
      p1[r] = exp2f(a1 - mn);
      lrow[r] = lrow[r] * alpha[r] + row16_sum(p0[r] + p1[r]);
    }
#pragma unroll
    for (int r = 0; r < 8; ++r) {
      o0[r] *= alpha[r]; o1[r] *= alpha[r];
      o2[r] *= alpha[r]; o3[r] *= alpha[r];
    }

    // ---- relayout P (C-frag f32) -> LDS row-major 16x32 -> A-frag f16 ----
    __syncthreads();
#pragma unroll
    for (int r = 0; r < 8; ++r) {
      pbuf[(mo + r) * 32 + n]      = (f16)p0[r];
      pbuf[(mo + r) * 32 + 16 + n] = (f16)p1[r];
    }
    __syncthreads();
    const v16h pa = load_frag_rowmajor(pbuf, 32, lane);

    // ---- P (16x32) * V_tile (32x64): B frags from Vt rows (d), k = kv ----
    {
      v16h vb0 = load_frag_rowmajor(vt,          S, lane);
      v16h vb1 = load_frag_rowmajor(vt + 16 * S, S, lane);
      v16h vb2 = load_frag_rowmajor(vt + 32 * S, S, lane);
      v16h vb3 = load_frag_rowmajor(vt + 48 * S, S, lane);
      o0 = wmma_f16(pa, vb0, o0);
      o1 = wmma_f16(pa, vb1, o1);
      o2 = wmma_f16(pa, vb2, o2);
      o3 = wmma_f16(pa, vb3, o3);
    }
  }

  // ---- epilogue: normalize and write y[s][h*64 + d] as f16 ----
  f16* yout = Y + (size_t)m0 * DMODEL + h * DHEAD;
#pragma unroll
  for (int r = 0; r < 8; ++r) {
    const float inv = 1.0f / lrow[r];
    const size_t row = (size_t)(mo + r);
    yout[row * DMODEL + n +  0] = (f16)(o0[r] * inv);
    yout[row * DMODEL + n + 16] = (f16)(o1[r] * inv);
    yout[row * DMODEL + n + 32] = (f16)(o2[r] * inv);
    yout[row * DMODEL + n + 48] = (f16)(o3[r] * inv);
  }
}

// ---------------------------------------------------------------------------
// Launch
// ---------------------------------------------------------------------------
extern "C" void kernel_launch(void* const* d_in, const int* in_sizes, int n_in,
                              void* d_out, int out_size, void* d_ws, size_t ws_size,
                              hipStream_t stream) {
  (void)in_sizes; (void)n_in; (void)out_size; (void)ws_size;

  const float* x  = (const float*)d_in[0];
  const float* Wq = (const float*)d_in[1];
  const float* bq = (const float*)d_in[2];
  const float* Wk = (const float*)d_in[3];
  const float* bk = (const float*)d_in[4];
  const float* Wv = (const float*)d_in[5];
  const float* bv = (const float*)d_in[6];
  const float* Wo = (const float*)d_in[7];
  const float* bo = (const float*)d_in[8];
  float* out = (float*)d_out;

  // workspace carve-up (~34.5 MB total)
  char* ws = (char*)d_ws;
  size_t off = 0;
  auto carve = [&](size_t bytes) {
    void* p = ws + off;
    off = (off + bytes + 255) & ~(size_t)255;
    return p;
  };
  f16* xh  = (f16*)carve((size_t)S_SEQ * DMODEL * sizeof(f16));
  f16* WqT = (f16*)carve((size_t)DMODEL * DMODEL * sizeof(f16));
  f16* WkT = (f16*)carve((size_t)DMODEL * DMODEL * sizeof(f16));
  f16* WvT = (f16*)carve((size_t)DMODEL * DMODEL * sizeof(f16));
  f16* WoT = (f16*)carve((size_t)DMODEL * DMODEL * sizeof(f16));
  f16* Qb  = (f16*)carve((size_t)NHEADS * S_SEQ * DHEAD * sizeof(f16));
  f16* Kb  = (f16*)carve((size_t)NHEADS * S_SEQ * DHEAD * sizeof(f16));
  f16* VbT = (f16*)carve((size_t)NHEADS * S_SEQ * DHEAD * sizeof(f16));
  f16* Yb  = (f16*)carve((size_t)S_SEQ * DMODEL * sizeof(f16));

  // 1) cast activations, transpose+cast weights
  {
    const int n = S_SEQ * DMODEL;
    cast_f32_to_f16<<<(n + 255) / 256, 256, 0, stream>>>(x, xh, n);
    dim3 tg(DMODEL / 32, DMODEL / 32), tb(32, 8);
    transpose_cast_f32_to_f16<<<tg, tb, 0, stream>>>(Wq, WqT, DMODEL, DMODEL);
    transpose_cast_f32_to_f16<<<tg, tb, 0, stream>>>(Wk, WkT, DMODEL, DMODEL);
    transpose_cast_f32_to_f16<<<tg, tb, 0, stream>>>(Wv, WvT, DMODEL, DMODEL);
    transpose_cast_f32_to_f16<<<tg, tb, 0, stream>>>(Wo, WoT, DMODEL, DMODEL);
  }

  // 2) QKV projections (Q,K head-major; V head-major transposed)
  {
    dim3 grid(DMODEL / 128, S_SEQ / 16);
    gemm_wmma<1><<<grid, 256, 0, stream>>>(xh, WqT, bq, Qb,  S_SEQ, DMODEL, DMODEL);
    gemm_wmma<1><<<grid, 256, 0, stream>>>(xh, WkT, bk, Kb,  S_SEQ, DMODEL, DMODEL);
    gemm_wmma<3><<<grid, 256, 0, stream>>>(xh, WvT, bv, VbT, S_SEQ, DMODEL, DMODEL);
  }

  // 3) flash attention, one wave per (head, 16-query tile)
  flash_attn_wmma<<<dim3(S_SEQ / 16, NHEADS), 32, 0, stream>>>(Qb, Kb, VbT, Yb, S_SEQ);

  // 4) output projection (f32 result straight into d_out)
  gemm_wmma<2><<<dim3(DMODEL / 128, S_SEQ / 16), 256, 0, stream>>>(
      Yb, WoT, bo, out, S_SEQ, DMODEL, DMODEL);
}
